// SymphoniesLayer_51161650430129
// MI455X (gfx1250) — compile-verified, matrix operand order checked by reading, not run on hardware
//
#include <hip/hip_runtime.h>
#include <hip/hip_bf16.h>

typedef __attribute__((ext_vector_type(16))) __bf16 v16bf;
typedef __attribute__((ext_vector_type(8)))  float  v8f;
typedef unsigned short ushort_t;
typedef unsigned int   uint32;

#define D_MODEL 256
#define NHEAD   8
#define NVOX    32768
#define NFOV_C  16384
#define NQ_C    100
#define DHID    1024
#define NIMG    10080

// ---------------------------------------------------------------------------
// fp32 -> bf16 round-to-nearest-even
__device__ __forceinline__ ushort_t f2bf(float f) {
  union { float f; uint32 u; } c; c.f = f;
  uint32 u = c.u;
  uint32 r = u + 0x7fffu + ((u >> 16) & 1u);
  return (ushort_t)(r >> 16);
}

// ---------------------------------------------------------------------------
// Generic GEMM: C[M,N] = act(A[M,K] * W[K,N] + bias), fp32 in/out,
// bf16 WMMA (v_wmma_f32_16x16x32_bf16) with f32 accumulate.
// Block tile 128x128 (8 waves), wave tile 32x64 = 2x4 WMMA tiles, K step 32.
// Staging is branchless: OOB rows/col-groups are clamped (their products land
// only in C rows/cols that the guarded epilogue never stores). Prefetch of the
// next A K-tile is branchless too (K offset clamped, prefetch is speculative).
#define BM 128
#define BN 128
#define BK 32
#define LDA 40   // ushort stride (32 + pad) -> 16B-aligned b128 fragment loads
#define LDB 40

__global__ __launch_bounds__(256)
void gemm_kernel(const float* __restrict__ A, const float* __restrict__ W,
                 const float* __restrict__ bias, float* __restrict__ C,
                 int M, int N, int K, int act)
{
  __shared__ ushort_t As[BM * LDA];
  __shared__ ushort_t Bs[BN * LDB];

  const int tid   = threadIdx.x;
  const int lane  = tid & 31;
  const int wave  = tid >> 5;
  const int lh    = lane >> 4;   // lane half selects K sub-blocks
  const int ll    = lane & 15;   // M (A) / N (B) index within 16x16 tile
  const int waveM = wave >> 1;   // 0..3 -> 32-row slice
  const int waveN = wave & 1;    // 0..1 -> 64-col slice
  const int m0 = blockIdx.x * BM;
  const int n0 = blockIdx.y * BN;

  v8f zero = {};
  v8f acc[2][4];
#pragma unroll
  for (int i = 0; i < 2; i++)
#pragma unroll
    for (int j = 0; j < 4; j++) acc[i][j] = zero;

  // A staging: 128 rows x 32 cols, thread -> (row tid/8 + 32i, col (tid%8)*4)
  const int arow = tid >> 3;
  const int acol = (tid & 7) * 4;
  // B staging: 32 rows x 128 cols, thread -> (k tid/32 + 8*kh, col (tid%32)*4)
  const int bk0 = tid >> 5;
  const int bn  = (tid & 31) * 4;
  const int bg  = min(n0 + bn, N - 4);   // clamped col group (N multiple of 4)

  for (int kk = 0; kk < K; kk += BK) {
    int kpf = min(kk + BK, K - BK);      // next-tile prefetch K offset, in-bounds
    // ---- stage A tile (fp32 -> bf16), branchless clamped rows ----
#pragma unroll
    for (int i = 0; i < 4; i++) {
      int r  = arow + i * 32;
      int gr = min(m0 + r, M - 1);
      const float* ap = A + (size_t)gr * K;
      float4 v = *(const float4*)(ap + kk + acol);
      __builtin_prefetch(ap + kpf + acol, 0, 1);
      union { ushort_t s[4]; uint2 u; } p;
      p.s[0] = f2bf(v.x); p.s[1] = f2bf(v.y); p.s[2] = f2bf(v.z); p.s[3] = f2bf(v.w);
      *(uint2*)(&As[r * LDA + acol]) = p.u;
    }
    // ---- stage B tile transposed into Bs[n][k], branchless ----
#pragma unroll
    for (int kh = 0; kh < 4; kh++) {
      int k = bk0 + kh * 8;
      float4 v = *(const float4*)(W + (size_t)(kk + k) * N + bg);
      Bs[(bn + 0) * LDB + k] = f2bf(v.x);
      Bs[(bn + 1) * LDB + k] = f2bf(v.y);
      Bs[(bn + 2) * LDB + k] = f2bf(v.z);
      Bs[(bn + 3) * LDB + k] = f2bf(v.w);
    }
    __syncthreads();

    union Frag { uint4 q[2]; v16bf v; };
    Frag af[2], bfg[4];
#pragma unroll
    for (int mt = 0; mt < 2; mt++) {
      const ushort_t* p = &As[(waveM * 32 + mt * 16 + ll) * LDA + lh * 8];
      af[mt].q[0] = *(const uint4*)(p);
      af[mt].q[1] = *(const uint4*)(p + 16);
    }
#pragma unroll
    for (int nt = 0; nt < 4; nt++) {
      const ushort_t* p = &Bs[(waveN * 64 + nt * 16 + ll) * LDB + lh * 8];
      bfg[nt].q[0] = *(const uint4*)(p);
      bfg[nt].q[1] = *(const uint4*)(p + 16);
    }
#pragma unroll
    for (int mt = 0; mt < 2; mt++)
#pragma unroll
      for (int nt = 0; nt < 4; nt++)
        acc[mt][nt] = __builtin_amdgcn_wmma_f32_16x16x32_bf16(
            false, af[mt].v, false, bfg[nt].v, (short)0, acc[mt][nt], false, false);
    __syncthreads();
  }

  // epilogue: D layout -> row = base + (lane/16)*8 + r, col = base + lane%16
#pragma unroll
  for (int mt = 0; mt < 2; mt++) {
#pragma unroll
    for (int nt = 0; nt < 4; nt++) {
      int gcol = n0 + waveN * 64 + nt * 16 + ll;
#pragma unroll
      for (int r = 0; r < 8; r++) {
        int grow = m0 + waveM * 32 + mt * 16 + lh * 8 + r;
        if (grow < M && gcol < N) {
          float v = acc[mt][nt][r];
          if (bias) v += bias[gcol];
          if (act == 1) {  // tanh-approx GELU (jax.nn.gelu default)
            float x = v;
            float t = 0.7978845608028654f * (x + 0.044715f * x * x * x);
            v = 0.5f * x * (1.0f + tanhf(t));
          }
          C[(size_t)grow * N + gcol] = v;
        }
      }
    }
  }
}

// ---------------------------------------------------------------------------
// Per-head attention with online softmax. K/V (<=128 keys) staged in LDS.
__global__ __launch_bounds__(256)
void attn_kernel(const float* __restrict__ Q, const float* __restrict__ Km,
                 const float* __restrict__ Vm, float* __restrict__ O,
                 int Nq, int Nk)
{
  __shared__ float Ks[128 * 32];
  __shared__ float Vs[128 * 32];
  const int h   = blockIdx.y;
  const int tid = threadIdx.x;
  for (int i = tid; i < Nk * 32; i += 256) {
    int k = i >> 5, d = i & 31;
    Ks[i] = Km[(size_t)k * D_MODEL + h * 32 + d];
    Vs[i] = Vm[(size_t)k * D_MODEL + h * 32 + d];
  }
  __syncthreads();
  int q = blockIdx.x * 256 + tid;
  if (q >= Nq) return;
  float qv[32], accv[32];
#pragma unroll
  for (int d = 0; d < 32; d++) { qv[d] = Q[(size_t)q * D_MODEL + h * 32 + d]; accv[d] = 0.f; }
  float m = -1e30f, l = 0.0f;
  const float scale = 0.17677669529663687f;  // 1/sqrt(32)
  for (int k = 0; k < Nk; k++) {
    float s = 0.0f;
#pragma unroll
    for (int d = 0; d < 32; d++) s += qv[d] * Ks[k * 32 + d];
    s *= scale;
    float mn = fmaxf(m, s);
    float f  = __expf(m - mn);
    float p  = __expf(s - mn);
    l = l * f + p;
#pragma unroll
    for (int d = 0; d < 32; d++) accv[d] = accv[d] * f + p * Vs[k * 32 + d];
    m = mn;
  }
  float inv = 1.0f / l;
#pragma unroll
  for (int d = 0; d < 32; d++) O[(size_t)q * D_MODEL + h * 32 + d] = accv[d] * inv;
}

// ---------------------------------------------------------------------------
// y[row] = LayerNorm(a[row] + b[row]) * gamma + beta, D=256, one block/row.
__global__ __launch_bounds__(256)
void add_ln_kernel(const float* __restrict__ a, const float* __restrict__ b,
                   const float* __restrict__ beta, const float* __restrict__ gamma,
                   float* __restrict__ y, int rows)
{
  __shared__ float red[256];
  int row = blockIdx.x;
  if (row >= rows) return;
  int c = threadIdx.x;
  float v = a[(size_t)row * 256 + c] + b[(size_t)row * 256 + c];
  red[c] = v;
  __syncthreads();
  for (int s = 128; s > 0; s >>= 1) { if (c < s) red[c] += red[c + s]; __syncthreads(); }
  float mean = red[0] * (1.0f / 256.0f);
  __syncthreads();
  float d = v - mean;
  red[c] = d * d;
  __syncthreads();
  for (int s = 128; s > 0; s >>= 1) { if (c < s) red[c] += red[c + s]; __syncthreads(); }
  float var = red[0] * (1.0f / 256.0f);
  y[(size_t)row * 256 + c] = d * rsqrtf(var + 1e-5f) * gamma[c] + beta[c];
}

// ---------------------------------------------------------------------------
// Transpose [256, ncols] -> [ncols, 256] via 32x32 LDS tiles.
__global__ __launch_bounds__(256)
void transpose_kernel(const float* __restrict__ src, float* __restrict__ dst, int ncols)
{
  __shared__ float tile[32][33];
  int bx = blockIdx.x * 32;       // token offset
  int by = blockIdx.y * 32;       // channel offset (0..224)
  int tx = threadIdx.x & 31;
  int ty = threadIdx.x >> 5;      // 0..7
#pragma unroll
  for (int j = 0; j < 4; j++) {
    int r = ty + j * 8, col = bx + tx;
    if (col < ncols) tile[r][tx] = src[(size_t)(by + r) * ncols + col];
  }
  __syncthreads();
#pragma unroll
  for (int j = 0; j < 4; j++) {
    int r = ty + j * 8, orow = bx + r;
    if (orow < ncols) dst[(size_t)orow * 256 + by + tx] = tile[tx][r];
  }
}

// ---------------------------------------------------------------------------
// FOV index extraction: single wave32, ballot-based stable compaction.
__global__ void fov_idx_kernel(const int* __restrict__ mask, int* __restrict__ idx,
                               int nvox, int nfov)
{
  int lane = threadIdx.x;
  int count = 0;
  for (int i = 0; i < nvox; i += 32) {
    int v = (i + lane < nvox) ? (mask[i + lane] != 0) : 0;
    unsigned bb = (unsigned)__ballot(v);
    int pos = count + __popc(bb & ((1u << lane) - 1u));
    if (v && pos < nfov) idx[pos] = i + lane;
    count += __popc(bb);
  }
}

__global__ void gather_fov_kernel(const float* __restrict__ scene, const int* __restrict__ idx,
                                  float* __restrict__ fov)
{
  int gid = blockIdx.x * 256 + threadIdx.x;   // NFOV*256 elems
  int n = gid >> 8, c = gid & 255;
  fov[(size_t)n * 256 + c] = scene[(size_t)c * NVOX + idx[n]];
}

__global__ void scatter_fov_kernel(const float* __restrict__ fov, const int* __restrict__ idx,
                                   float* __restrict__ scene)
{
  int gid = blockIdx.x * 256 + threadIdx.x;
  int n = gid >> 8, c = gid & 255;
  scene[(size_t)c * NVOX + idx[n]] = fov[(size_t)n * 256 + c];
}

// softmax over LP contiguous values per (row, head): layout [rows, H*LP]
__global__ void softmax_lp_kernel(float* __restrict__ aw, int rows, int LP)
{
  int gid = blockIdx.x * 256 + threadIdx.x;
  if (gid >= rows * NHEAD) return;
  float* p = aw + (size_t)gid * LP;
  float mx = -1e30f;
  for (int i = 0; i < LP; i++) mx = fmaxf(mx, p[i]);
  float s = 0.0f;
  for (int i = 0; i < LP; i++) { float e = __expf(p[i] - mx); p[i] = e; s += e; }
  float inv = 1.0f / s;
  for (int i = 0; i < LP; i++) p[i] *= inv;
}

// ---------------------------------------------------------------------------
// 3D deformable sampling: one wave per (query, head), lane = channel (Dh=32).
// L=1, Pn=8, shape (64,64,8), strides (512,8,1).
__global__ __launch_bounds__(256)
void deform3d_kernel(const float* __restrict__ vproj, const float* __restrict__ off,
                     const float* __restrict__ aw, const float* __restrict__ ref,
                     const int* __restrict__ idx, float* __restrict__ out, int nq)
{
  int pair = blockIdx.x * 8 + (threadIdx.x >> 5);
  int lane = threadIdx.x & 31;
  if (pair >= nq * NHEAD) return;
  int n = pair >> 3, h = pair & 7;
  int rr = idx ? idx[n] : n;
  float r0 = ref[(size_t)rr * 3 + 0];
  float r1 = ref[(size_t)rr * 3 + 1];
  float r2 = ref[(size_t)rr * 3 + 2];
  float acc = 0.0f;
#pragma unroll
  for (int p = 0; p < 8; p++) {
    const float* ob = off + (size_t)n * 192 + (h * 8 + p) * 3;
    float cx = (r0 + ob[0] * (1.0f / 64.0f)) * 64.0f - 0.5f;
    float cy = (r1 + ob[1] * (1.0f / 64.0f)) * 64.0f - 0.5f;
    float cz = (r2 + ob[2] * (1.0f / 8.0f))  * 8.0f  - 0.5f;
    float fx = floorf(cx), fy = floorf(cy), fz = floorf(cz);
    float wp = aw[(size_t)n * 64 + h * 8 + p];
    float sp = 0.0f;
#pragma unroll
    for (int cc = 0; cc < 8; cc++) {
      float ix = fx + (cc & 1);
      float iy = fy + ((cc >> 1) & 1);
      float iz = fz + (cc >> 2);
      float w = (1.0f - fabsf(cx - ix)) * (1.0f - fabsf(cy - iy)) * (1.0f - fabsf(cz - iz));
      bool valid = (ix >= 0.f && ix < 64.f && iy >= 0.f && iy < 64.f && iz >= 0.f && iz < 8.f);
      int xi = min(max((int)ix, 0), 63);
      int yi = min(max((int)iy, 0), 63);
      int zi = min(max((int)iz, 0), 7);
      float g = vproj[(size_t)(xi * 512 + yi * 8 + zi) * 256 + h * 32 + lane];
      sp += valid ? w * g : 0.0f;
    }
    acc += wp * sp;
  }
  out[(size_t)n * 256 + h * 32 + lane] = acc;
}

// 2D multi-level deformable sampling: L=3 {(48,160),(24,80),(12,40)}, Pn=4.
__global__ __launch_bounds__(256)
void deform2d_kernel(const float* __restrict__ vproj, const float* __restrict__ off,
                     const float* __restrict__ aw, const float* __restrict__ ref2,
                     float* __restrict__ out, int nq)
{
  int pair = blockIdx.x * 8 + (threadIdx.x >> 5);
  int lane = threadIdx.x & 31;
  if (pair >= nq * NHEAD) return;
  int n = pair >> 3, h = pair & 7;
  const int Hs[3] = {48, 24, 12};
  const int Ws[3] = {160, 80, 40};
  const int Ls[3] = {0, 7680, 9600};
  float acc = 0.0f;
  for (int l = 0; l < 3; l++) {
    float Wf = (float)Ws[l], Hf = (float)Hs[l];
    float rx = ref2[(size_t)n * 6 + l * 2 + 0];
    float ry = ref2[(size_t)n * 6 + l * 2 + 1];
    for (int p = 0; p < 4; p++) {
      const float* ob = off + (size_t)n * 192 + ((h * 3 + l) * 4 + p) * 2;
      float cx = (rx + ob[0] / Wf) * Wf - 0.5f;
      float cy = (ry + ob[1] / Hf) * Hf - 0.5f;
      float fx = floorf(cx), fy = floorf(cy);
      float wp = aw[(size_t)n * 96 + h * 12 + l * 4 + p];
      float sp = 0.0f;
      for (int cc = 0; cc < 4; cc++) {
        float ix = fx + (cc & 1);
        float iy = fy + (cc >> 1);
        float w = (1.0f - fabsf(cx - ix)) * (1.0f - fabsf(cy - iy));
        bool valid = (ix >= 0.f && ix < Wf && iy >= 0.f && iy < Hf);
        int xi = min(max((int)ix, 0), Ws[l] - 1);
        int yi = min(max((int)iy, 0), Hs[l] - 1);
        float g = vproj[(size_t)(Ls[l] + xi + yi * Ws[l]) * 256 + h * 32 + lane];
        sp += valid ? w * g : 0.0f;
      }
      acc += wp * sp;
    }
  }
  out[(size_t)n * 256 + h * 32 + lane] = acc;
}

// ---------------------------------------------------------------------------
extern "C" void kernel_launch(void* const* d_in, const int* in_sizes, int n_in,
                              void* d_out, int out_size, void* d_ws, size_t ws_size,
                              hipStream_t stream) {
  (void)in_sizes; (void)n_in; (void)out_size; (void)ws_size;
  // ---- inputs (top-level dict in insertion order; params = pytree leaves,
  //      sorted keys DFS: cross_attn, query_image_deform, query_scene_deform,
  //      scene_deform, self_attn; lin {W,b}; ln {b,g}) ----
  const float* scene_embed = (const float*)d_in[0];
  const float* inst_q      = (const float*)d_in[1];
  const float* feat0       = (const float*)d_in[2];
  const float* feat1       = (const float*)d_in[3];
  const float* feat2       = (const float*)d_in[4];
  const float* ref_2d      = (const float*)d_in[5];
  const float* ref_3d      = (const float*)d_in[6];
  const float* ref_vox     = (const float*)d_in[7];
  const int*   fov_mask    = (const int*)d_in[8];
  auto P = [&](int i) { return (const float*)d_in[i]; };
  // cross_attn: k 10/11, o 12/13, q 14/15, v 16/17, ln1 b18 g19, ln2 b20 g21,
  //             fc1 22/23, fc2 24/25
  // query_image_deform: aw 26/27, ln1 28/29, ln2 30/31, fc1 32/33, fc2 34/35,
  //             off 36/37, out 38/39, vproj 40/41
  // query_scene_deform: 42..57 (same pattern)
  // scene_deform: 58..73
  // self_attn: k 74/75, o 76/77, q 78/79, v 80/81, ln1 82/83, ln2 84/85,
  //            fc1 86/87, fc2 88/89

  float* out = (float*)d_out;
  float* scene_out = out;                       // 256*NVOX
  float* iq_out    = out + (size_t)NVOX * 256;  // 100*256

  // ---- workspace layout ----
  size_t o = 0;
  auto alloc = [&](size_t bytes) { size_t cur = o; o += (bytes + 255) & ~(size_t)255; return cur; };
  char* ws = (char*)d_ws;
  int*   idx       = (int*)(ws + alloc((size_t)NFOV_C * 4));
  float* iq0       = (float*)(ws + alloc((size_t)NQ_C * 256 * 4));
  float* iq1       = (float*)(ws + alloc((size_t)NQ_C * 256 * 4));
  float* iq2       = (float*)(ws + alloc((size_t)NQ_C * 256 * 4));
  float* iqh       = (float*)(ws + alloc((size_t)NQ_C * DHID * 4));
  float* kvk       = (float*)(ws + alloc((size_t)NQ_C * 256 * 4));
  float* kvv       = (float*)(ws + alloc((size_t)NQ_C * 256 * 4));
  float* off_iq    = (float*)(ws + alloc((size_t)NQ_C * 192 * 4));
  float* aw_iq     = (float*)(ws + alloc((size_t)NQ_C * 96 * 4));
  float* q_fov     = (float*)(ws + alloc((size_t)NFOV_C * 256 * 4));
  float* t0        = (float*)(ws + alloc((size_t)NFOV_C * 256 * 4));
  float* t1        = (float*)(ws + alloc((size_t)NFOV_C * 256 * 4));
  float* h_fov     = (float*)(ws + alloc((size_t)NFOV_C * DHID * 4));
  float* off_fov   = (float*)(ws + alloc((size_t)NFOV_C * 192 * 4));
  float* aw_fov    = (float*)(ws + alloc((size_t)NFOV_C * 64 * 4));
  float* scene_tok = (float*)(ws + alloc((size_t)NVOX * 256 * 4));
  float* vscene    = (float*)(ws + alloc((size_t)NVOX * 256 * 4));
  float* feats_fl  = (float*)(ws + alloc((size_t)NIMG * 256 * 4));
  float* vimg      = (float*)(ws + alloc((size_t)NIMG * 256 * 4));

  auto gemm = [&](const float* A, int Wi, int Bi, float* Cp, int M, int N, int K, int act) {
    dim3 g((M + BM - 1) / BM, (N + BN - 1) / BN);
    gemm_kernel<<<g, 256, 0, stream>>>(A, P(Wi), Bi >= 0 ? P(Bi) : nullptr, Cp, M, N, K, act);
  };
  auto addln = [&](const float* a, const float* b, int Bi, int Gi, float* y, int rows) {
    add_ln_kernel<<<rows, 256, 0, stream>>>(a, b, P(Bi), P(Gi), y, rows);
  };

  // ---- 0. fov indices ----
  fov_idx_kernel<<<1, 32, 0, stream>>>(fov_mask, idx, NVOX, NFOV_C);

  // ---- 1. iq = tlayer(self_attn, inst, inst, inst) ----
  gemm(inst_q, 78, 79, iq1, NQ_C, 256, 256, 0);           // q proj
  gemm(inst_q, 74, 75, kvk, NQ_C, 256, 256, 0);           // k proj
  gemm(inst_q, 80, 81, kvv, NQ_C, 256, 256, 0);           // v proj
  attn_kernel<<<dim3(1, NHEAD), 256, 0, stream>>>(iq1, kvk, kvv, iq2, NQ_C, NQ_C);
  gemm(iq2, 76, 77, iq1, NQ_C, 256, 256, 0);              // o proj
  addln(iq1, inst_q, 82, 83, iq0, NQ_C);                  // x = LN(q + mha)
  gemm(iq0, 86, 87, iqh, NQ_C, DHID, 256, 1);             // fc1 + gelu
  gemm(iqh, 88, 89, iq1, NQ_C, 256, DHID, 0);             // fc2
  addln(iq1, iq0, 84, 85, iq0, NQ_C);                     // iq

  // ---- 2. fov tokens + cross attention ----
  gather_fov_kernel<<<NFOV_C, 256, 0, stream>>>(scene_embed, idx, q_fov);
  gemm(q_fov, 14, 15, t0, NFOV_C, 256, 256, 0);           // q proj
  gemm(iq0, 10, 11, kvk, NQ_C, 256, 256, 0);              // k proj
  gemm(iq0, 16, 17, kvv, NQ_C, 256, 256, 0);              // v proj
  attn_kernel<<<dim3(NFOV_C / 256, NHEAD), 256, 0, stream>>>(t0, kvk, kvv, t1, NFOV_C, NQ_C);
  gemm(t1, 12, 13, t0, NFOV_C, 256, 256, 0);              // o proj
  addln(t0, q_fov, 18, 19, q_fov, NFOV_C);
  gemm(q_fov, 22, 23, h_fov, NFOV_C, DHID, 256, 1);
  gemm(h_fov, 24, 25, t0, NFOV_C, 256, DHID, 0);
  addln(t0, q_fov, 20, 21, q_fov, NFOV_C);

  // ---- 3. scene_deform dlayer on fov (value = original scene tokens) ----
  transpose_kernel<<<dim3(NVOX / 32, 8), 256, 0, stream>>>(scene_embed, scene_tok, NVOX);
  gemm(scene_tok, 72, 73, vscene, NVOX, 256, 256, 0);     // vproj
  gemm(q_fov, 68, 69, off_fov, NFOV_C, 192, 256, 0);      // offsets
  gemm(q_fov, 58, 59, aw_fov, NFOV_C, 64, 256, 0);        // attn weights
  softmax_lp_kernel<<<(NFOV_C * NHEAD + 255) / 256, 256, 0, stream>>>(aw_fov, NFOV_C, 8);
  deform3d_kernel<<<(NFOV_C * NHEAD) / 8, 256, 0, stream>>>(vscene, off_fov, aw_fov,
                                                            ref_vox, idx, t1, NFOV_C);
  gemm(t1, 70, 71, t0, NFOV_C, 256, 256, 0);              // out proj
  addln(t0, q_fov, 60, 61, q_fov, NFOV_C);
  gemm(q_fov, 64, 65, h_fov, NFOV_C, DHID, 256, 1);
  gemm(h_fov, 66, 67, t0, NFOV_C, 256, DHID, 0);
  addln(t0, q_fov, 62, 63, q_fov, NFOV_C);

  // ---- 4. scene output = scene_embed with fov columns replaced ----
  hipMemcpyAsync(scene_out, scene_embed, (size_t)NVOX * 256 * 4,
                 hipMemcpyDeviceToDevice, stream);
  scatter_fov_kernel<<<NFOV_C, 256, 0, stream>>>(q_fov, idx, scene_out);

  // ---- 5. query_scene_deform on iq (value = updated scene tokens) ----
  transpose_kernel<<<dim3(NVOX / 32, 8), 256, 0, stream>>>(scene_out, scene_tok, NVOX);
  gemm(scene_tok, 56, 57, vscene, NVOX, 256, 256, 0);
  gemm(iq0, 52, 53, off_iq, NQ_C, 192, 256, 0);
  gemm(iq0, 42, 43, aw_iq, NQ_C, 64, 256, 0);
  softmax_lp_kernel<<<(NQ_C * NHEAD + 255) / 256, 256, 0, stream>>>(aw_iq, NQ_C, 8);
  deform3d_kernel<<<(NQ_C * NHEAD + 7) / 8, 256, 0, stream>>>(vscene, off_iq, aw_iq,
                                                              ref_3d, nullptr, iq2, NQ_C);
  gemm(iq2, 54, 55, iq1, NQ_C, 256, 256, 0);
  addln(iq1, iq0, 44, 45, iq0, NQ_C);
  gemm(iq0, 48, 49, iqh, NQ_C, DHID, 256, 1);
  gemm(iqh, 50, 51, iq1, NQ_C, 256, DHID, 0);
  addln(iq1, iq0, 46, 47, iq0, NQ_C);

  // ---- 6. flatten image feats ----
  transpose_kernel<<<dim3(7680 / 32, 8), 256, 0, stream>>>(feat0, feats_fl, 7680);
  transpose_kernel<<<dim3(1920 / 32, 8), 256, 0, stream>>>(feat1, feats_fl + (size_t)7680 * 256, 1920);
  transpose_kernel<<<dim3(480 / 32, 8), 256, 0, stream>>>(feat2, feats_fl + (size_t)9600 * 256, 480);

  // ---- 7. query_image_deform on iq ----
  gemm(feats_fl, 40, 41, vimg, NIMG, 256, 256, 0);
  gemm(iq0, 36, 37, off_iq, NQ_C, 192, 256, 0);
  gemm(iq0, 26, 27, aw_iq, NQ_C, 96, 256, 0);
  softmax_lp_kernel<<<(NQ_C * NHEAD + 255) / 256, 256, 0, stream>>>(aw_iq, NQ_C, 12);
  deform2d_kernel<<<(NQ_C * NHEAD + 7) / 8, 256, 0, stream>>>(vimg, off_iq, aw_iq,
                                                              ref_2d, iq2, NQ_C);
  gemm(iq2, 38, 39, iq1, NQ_C, 256, 256, 0);
  addln(iq1, iq0, 28, 29, iq0, NQ_C);
  gemm(iq0, 32, 33, iqh, NQ_C, DHID, 256, 1);
  gemm(iqh, 34, 35, iq1, NQ_C, 256, DHID, 0);
  addln(iq1, iq0, 30, 31, iq_out, NQ_C);                  // final iq -> d_out
}
